// FlowSimCorrespondenceGenerationArch_3040836846257
// MI455X (gfx1250) — matrix operand order repaired and checked by reading, not compiled, not run on hardware
//
#include <hip/hip_runtime.h>
#include <hip/hip_bf16.h>

// ---------------------------------------------------------------------------
// FlowSim correspondence generation for MI455X (gfx1250, wave32, WMMA).
//
//  (1) corr = pin(3844x2304) @ pref^T per batch (B=4), ~272 GFLOP total
//      -> v_wmma_f32_16x16x32_f16 with streaming argmax (corr never stored).
//      Operands (~146MB padded f16) are L2-resident (192MB): block-tile
//      128x128, B staged in double-buffered LDS via async global->LDS
//      copies, 8 accumulators/wave -> AI = 64 FLOP/B vs L2.
//      B fragments are software-pipelined 1-deep so ds_load latency hides
//      under the previous WMMA (partial s_wait_dscnt instead of 0).
//  (2) VGG prefix convs (~60 GFLOP) -> f32 FMA, thread-per-output.
//  (3) flow/off/sim fan-out -> elementwise, trivial vs 23.3 TB/s HBM.
// ---------------------------------------------------------------------------

#define B_N    4
#define C_F    256
#define HW     64
#define PW     62           // 64 - 3 + 1
#define NP     3844         // 62*62 patches
#define NP_PAD 3968         // 31 * 128 (block tile), zero-padded rows
#define KD     2304         // 256 * 9
#define KC     64           // K halves staged per LDS buffer
#define NK     (KD / KC)    // 36 k-chunks (even -> clean double buffer)

typedef __attribute__((ext_vector_type(16))) _Float16 v16h;
typedef __attribute__((ext_vector_type(8)))  _Float16 v8h;
typedef __attribute__((ext_vector_type(8)))  float    v8f;
typedef __attribute__((ext_vector_type(4)))  int      v4i;

// ---- gfx1250 async global->LDS copy (guarded; fallback = load + ds_store) --
#if defined(__has_builtin)
#  if __has_builtin(__builtin_amdgcn_global_load_async_to_lds_b128)
#    define HAVE_ASYNC_COPY 1
#  endif
#endif
#ifndef HAVE_ASYNC_COPY
#  define HAVE_ASYNC_COPY 0
#endif

__device__ __forceinline__ void async_copy16(const void* g, void* l) {
#if HAVE_ASYNC_COPY
    // prototype: void(v4i addrspace(1)*, v4i addrspace(3)*, imm offset, imm cpol)
    __builtin_amdgcn_global_load_async_to_lds_b128(
        (__attribute__((address_space(1))) v4i*)(v4i*)const_cast<void*>(g),
        (__attribute__((address_space(3))) v4i*)(v4i*)l, 0, 0);
#else
    *(v8h*)l = *(const v8h*)g;
#endif
}

__device__ __forceinline__ void async_wait_all() {
#if HAVE_ASYNC_COPY
#  if __has_builtin(__builtin_amdgcn_s_wait_asynccnt)
    __builtin_amdgcn_s_wait_asynccnt(0);
#  else
    asm volatile("s_wait_asynccnt 0" ::: "memory");
#  endif
#endif
}

// ------------------------- per-pixel channel norms -------------------------
__global__ void pixnorm_kernel(const float* __restrict__ f, float* __restrict__ invn) {
    int t = blockIdx.x * blockDim.x + threadIdx.x;
    if (t >= B_N * HW * HW) return;
    int b = t >> 12, pix = t & 4095;
    const float* p = f + (size_t)b * C_F * HW * HW + pix;
    float s = 0.f;
#pragma unroll 8
    for (int c = 0; c < C_F; ++c) { float v = p[c * HW * HW]; s += v * v; }
    invn[t] = 1.0f / fmaxf(sqrtf(s), 1e-12f);
}

// --------------- patch matrix build (one wave per patch row) ---------------
__global__ void __launch_bounds__(256)
build_patch_kernel(const float* __restrict__ feat, const float* __restrict__ invn,
                   _Float16* __restrict__ dst, float* __restrict__ row_scale,
                   int normalize_rows) {
    int lane = threadIdx.x & 31;
    int row = blockIdx.x * 8 + (threadIdx.x >> 5);
    int b = blockIdx.y;
    _Float16* drow = dst + ((size_t)b * NP_PAD + row) * KD;
    if (row >= NP) {                       // zero pad rows so WMMA reads 0
        for (int i = lane; i < KD; i += 32) drow[i] = (_Float16)0.0f;
        return;
    }
    int y = row / PW, x = row % PW;
    const float* fb = feat + (size_t)b * C_F * HW * HW;
    const float* nb = invn + b * HW * HW;
    float ss = 0.f;
    for (int i = lane; i < KD; i += 32) {
        int c = i / 9, s9 = i % 9, dy = s9 / 3, dx = s9 % 3;
        int yy = y + dy, xx = x + dx;
        float v = fb[(c * HW + yy) * HW + xx] * nb[yy * HW + xx];
        ss += v * v;
        drow[i] = (_Float16)v;
    }
#pragma unroll
    for (int o = 16; o >= 1; o >>= 1) ss += __shfl_xor(ss, o, 32);
    float inv = 1.0f / (sqrtf(ss) + 1e-5f);
    if (normalize_rows) {
        for (int i = lane; i < KD; i += 32)
            drow[i] = (_Float16)((float)drow[i] * inv);
    } else if (row_scale != nullptr && lane == 0) {
        row_scale[b * NP + row] = inv;
    }
}

// --------------- WMMA GEMM with streaming argmax over columns --------------
// Block = 8 waves, tile M=128 (16 rows/wave) x N-window 128 (8 acc/wave).
// B tile (128 rows x KC halves) double-buffered in LDS, filled with async
// global->LDS copies; A fragments come straight from global (L2 resident).
__device__ __forceinline__ void stage_b_tile(_Float16* __restrict__ dst,
                                             const _Float16* __restrict__ src, int t) {
    // 128 rows x KC(=64) halves = 16KB; 1024 16B-chunks over 256 threads.
#pragma unroll
    for (int i = 0; i < 4; ++i) {
        int c = t + i * 256;
        int row = c >> 3, seg = c & 7;
        const _Float16* g = src + (size_t)row * KD + seg * 8;
        _Float16* l = dst + row * KC + seg * 8;
        async_copy16(g, l);
    }
}

union Frag { v16h v; v8h h[2]; };

__global__ void __launch_bounds__(256)
corr_argmax_kernel(const _Float16* __restrict__ pin, const _Float16* __restrict__ pref,
                   const float* __restrict__ rscale,
                   int* __restrict__ oidx, float* __restrict__ oval) {
    __shared__ __align__(32) _Float16 ldsb[2][128 * KC];

    const int t = threadIdx.x;
    const int lane = t & 31, wave = t >> 5;
    const int r = lane & 15, half = lane >> 4;
    const int bb = blockIdx.y;
    const int m0 = blockIdx.x * 128 + wave * 16;

    const _Float16* A  = pin  + ((size_t)bb * NP_PAD + (m0 + r)) * KD;
    const _Float16* Pb = pref +  (size_t)bb * NP_PAD * KD;

    float best[8]; int bidx[8];
#pragma unroll
    for (int i = 0; i < 8; ++i) { best[i] = -3.0e38f; bidx[i] = 0; }

    const v8f zero = {0.f,0.f,0.f,0.f,0.f,0.f,0.f,0.f};
    for (int n0 = 0; n0 < NP_PAD; n0 += 128) {
        const _Float16* Bn = Pb + (size_t)n0 * KD;
        v8f acc[8];
#pragma unroll
        for (int j = 0; j < 8; ++j) acc[j] = zero;

        stage_b_tile(&ldsb[0][0], Bn, t);                 // prologue: k = 0
        int buf = 0;
        for (int kc = 0; kc < NK; ++kc) {
            const int k = kc * KC;
            async_wait_all();
            __syncthreads();                              // staged tile visible
            if (kc + 1 < NK)
                stage_b_tile(&ldsb[buf ^ 1][0], Bn + (k + KC), t);
            const _Float16* L = &ldsb[buf][0];

            // hoist both A fragments of this K-chunk (kk = 0 and 32)
            Frag a0, a1;
            a0.h[0] = *(const v8h*)(A + k +      16 * half);
            a0.h[1] = *(const v8h*)(A + k +  8 + 16 * half);
            a1.h[0] = *(const v8h*)(A + k + 32 + 16 * half);
            a1.h[1] = *(const v8h*)(A + k + 40 + 16 * half);

            // 16 fragments (8 n-subtiles x 2 kk), software-pipelined 1-deep
            const _Float16* lrow = L + (size_t)r * KC + 16 * half;
            Frag bcur, bnxt;
            bcur.h[0] = *(const v8h*)(lrow);
            bcur.h[1] = *(const v8h*)(lrow + 8);
#pragma unroll
            for (int u = 0; u < 16; ++u) {
                const int j  = u & 7;
                if (u < 15) {
                    const int un = u + 1;
                    const _Float16* lp = lrow + (size_t)(16 * (un & 7)) * KC
                                              + ((un >> 3) * 32);
                    bnxt.h[0] = *(const v8h*)(lp);
                    bnxt.h[1] = *(const v8h*)(lp + 8);
                }
                acc[j] = __builtin_amdgcn_wmma_f32_16x16x32_f16(
                    false, (u < 8) ? a0.v : a1.v, false, bcur.v,
                    (short)0, acc[j], false, false);
                bcur = bnxt;                               // renamed away (unrolled)
            }
            __syncthreads();                              // tile consumed
            buf ^= 1;
        }
        // streaming argmax update: lane column n = n0 + 16*j + r
#pragma unroll
        for (int j = 0; j < 8; ++j) {
            int n = n0 + 16 * j + r;
            if (n < NP) {
#pragma unroll
                for (int i = 0; i < 8; ++i)
                    if (acc[j][i] > best[i]) { best[i] = acc[j][i]; bidx[i] = n; }
            }
        }
    }
    // merge the 16 disjoint n-subsets of this half; first-index tie-break
#pragma unroll
    for (int o = 8; o >= 1; o >>= 1) {
#pragma unroll
        for (int i = 0; i < 8; ++i) {
            float ov = __shfl_xor(best[i], o, 32);
            int   oi = __shfl_xor(bidx[i], o, 32);
            if (ov > best[i] || (ov == best[i] && oi < bidx[i])) { best[i] = ov; bidx[i] = oi; }
        }
    }
    if (r == 0) {
#pragma unroll
        for (int i = 0; i < 8; ++i) {
            int m = m0 + half * 8 + i;
            if (m < NP) {
                oidx[bb * NP + m] = bidx[i];
                oval[bb * NP + m] = best[i] * rscale[bb * NP + m];  // norm_input
            }
        }
    }
}

// ----------------------- flow / offsets / sim fan-out ----------------------
__device__ inline void flow3_at(const int* __restrict__ ib, int yy, int xx,
                                float& fx, float& fy) {
    if (yy < PW && xx < PW) {
        int id = ib[yy * PW + xx];
        fx = (float)(id % PW) - (float)xx;
        fy = (float)(id / PW) - (float)yy;
    } else { fx = 0.f; fy = 0.f; }
}

__global__ void flow_out_kernel(const int* __restrict__ midx, const float* __restrict__ mval,
                                float* __restrict__ off_o, float* __restrict__ flow_o,
                                float* __restrict__ sim_o, int s) {
    int H = HW * s;
    int t = blockIdx.x * blockDim.x + threadIdx.x;
    if (t >= B_N * H * H) return;
    int x = t % H, y = (t / H) % H, b = t / (H * H);
    const int* ib = midx + b * NP;
    int ys = y / s, xs = x / s;
    float fs = (float)s;

    float fx, fy;
    flow3_at(ib, ys, xs, fx, fy);
    size_t fo = ((size_t)(b * H + y) * H + x) * 2;
    flow_o[fo] = fs * fx; flow_o[fo + 1] = fs * fy;

    float sv = 0.f;
    if (ys >= 1 && ys <= PW && xs >= 1 && xs <= PW)
        sv = mval[b * NP + (ys - 1) * PW + (xs - 1)];
    sim_o[(size_t)(b * H + y) * H + x] = sv;

#pragma unroll
    for (int i = 0; i < 3; ++i)
#pragma unroll
        for (int j = 0; j < 3; ++j) {
            float ox = 0.f, oy = 0.f;
            if (ys >= i && xs >= j) {
                float gx, gy;
                flow3_at(ib, ys - i, xs - j, gx, gy);
                ox = fs * gx; oy = fs * gy;
            }
            size_t oo = (((size_t)(b * 9 + i * 3 + j) * H + y) * H + x) * 2;
            off_o[oo] = ox; off_o[oo + 1] = oy;
        }
}

// -------------------------------- VGG prefix -------------------------------
__global__ void imgnorm_kernel(const float* __restrict__ img, float* __restrict__ out) {
    int t = blockIdx.x * blockDim.x + threadIdx.x;
    if (t >= B_N * 3 * 256 * 256) return;
    int c = (t >> 16) % 3;
    float m = (c == 0) ? 0.485f : (c == 1) ? 0.456f : 0.406f;
    float sd = (c == 0) ? 0.229f : (c == 1) ? 0.224f : 0.225f;
    out[t] = (img[t] - m) / sd;
}

template <int CIN>
__global__ void conv3x3_relu_kernel(const float* __restrict__ in, const float* __restrict__ w,
                                    const float* __restrict__ bias, float* __restrict__ out,
                                    int Cout, int H) {
    int t = blockIdx.x * blockDim.x + threadIdx.x;
    int total = B_N * Cout * H * H;
    if (t >= total) return;
    int x = t % H, y = (t / H) % H, o = (t / (H * H)) % Cout, b = t / (H * H * Cout);
    const float* wp = w + (size_t)o * CIN * 9;
    const float* ip = in + (size_t)b * CIN * H * H;
    float acc = bias[o];
    int y0 = y - 1, x0 = x - 1;
#pragma unroll 4
    for (int ci = 0; ci < CIN; ++ci) {
        const float* base = ip + (size_t)ci * H * H;
#pragma unroll
        for (int dy = 0; dy < 3; ++dy) {
            int yy = y0 + dy;
            if (yy < 0 || yy >= H) continue;
            const float* rp = base + (size_t)yy * H;
#pragma unroll
            for (int dx = 0; dx < 3; ++dx) {
                int xx = x0 + dx;
                if (xx < 0 || xx >= H) continue;
                acc = fmaf(rp[xx], wp[ci * 9 + dy * 3 + dx], acc);
            }
        }
    }
    out[t] = fmaxf(acc, 0.f);
}

__global__ void pool2x2_kernel(const float* __restrict__ in, float* __restrict__ out,
                               int C, int Hout) {
    int t = blockIdx.x * blockDim.x + threadIdx.x;
    if (t >= B_N * C * Hout * Hout) return;
    int x = t % Hout, y = (t / Hout) % Hout, c = (t / (Hout * Hout)) % C,
        b = t / (Hout * Hout * C);
    int Hin = Hout * 2;
    const float* p = in + (((size_t)b * C + c) * Hin + 2 * y) * Hin + 2 * x;
    out[t] = fmaxf(fmaxf(p[0], p[1]), fmaxf(p[Hin], p[Hin + 1]));
}

// --------------------------------- launch ----------------------------------
extern "C" void kernel_launch(void* const* d_in, const int* in_sizes, int n_in,
                              void* d_out, int out_size, void* d_ws, size_t ws_size,
                              hipStream_t stream) {
    (void)in_sizes; (void)n_in; (void)out_size; (void)ws_size;
    const float* f1  = (const float*)d_in[0];
    const float* f2  = (const float*)d_in[1];
    const float* img = (const float*)d_in[2];
    const float* w11 = (const float*)d_in[3];  const float* b11 = (const float*)d_in[4];
    const float* w12 = (const float*)d_in[5];  const float* b12 = (const float*)d_in[6];
    const float* w21 = (const float*)d_in[7];  const float* b21 = (const float*)d_in[8];
    const float* w22 = (const float*)d_in[9];  const float* b22 = (const float*)d_in[10];
    const float* w31 = (const float*)d_in[11]; const float* b31 = (const float*)d_in[12];
    float* out = (float*)d_out;
    char*  ws  = (char*)d_ws;

    // workspace layout (bytes)
    const size_t S_P     = (size_t)B_N * NP_PAD * KD * sizeof(_Float16); // 73,138,176
    const size_t o_pin   = 0;
    const size_t o_pref  = o_pin + S_P;
    const size_t o_norm1 = o_pref + S_P;
    const size_t o_norm2 = o_norm1 + 65536;
    const size_t o_pinn  = o_norm2 + 65536;
    const size_t o_idx   = o_pinn + 65536;
    const size_t o_val   = o_idx + 65536;

    _Float16* pin_h  = (_Float16*)(ws + o_pin);
    _Float16* pref_h = (_Float16*)(ws + o_pref);
    float* norm1 = (float*)(ws + o_norm1);
    float* norm2 = (float*)(ws + o_norm2);
    float* pinn  = (float*)(ws + o_pinn);
    int*   idx   = (int*)(ws + o_idx);
    float* val   = (float*)(ws + o_val);
    // VGG scratch aliases (valid after the corr GEMM consumed pin/pref)
    float* vt    = (float*)(ws + o_pin);                        // 67.1 MB tmp
    float* vp    = (float*)(ws + o_pref);                       // pool tmp
    float* vx    = (float*)(ws + o_pref + ((size_t)32 << 20));  // img norm

    // output layout (floats, reference tuple order)
    const size_t o_off3 = 0,        o_off2 = 294912,   o_off1 = 1474560;
    const size_t o_fl3  = 6193152,  o_fl2  = 6225920,  o_fl1  = 6356992;
    const size_t o_s3   = 6881280,  o_s2   = 6897664,  o_s1   = 6963200;
    const size_t o_r1   = 7225344,  o_r2   = 24002560, o_r3   = 32391168;

    // (1) matching path
    pixnorm_kernel<<<(B_N * HW * HW + 255) / 256, 256, 0, stream>>>(f1, norm1);
    pixnorm_kernel<<<(B_N * HW * HW + 255) / 256, 256, 0, stream>>>(f2, norm2);
    build_patch_kernel<<<dim3(NP_PAD / 8, B_N), 256, 0, stream>>>(f1, norm1, pin_h, pinn, 0);
    build_patch_kernel<<<dim3(NP_PAD / 8, B_N), 256, 0, stream>>>(f2, norm2, pref_h, nullptr, 1);
    corr_argmax_kernel<<<dim3(NP_PAD / 128, B_N), 256, 0, stream>>>(pin_h, pref_h, pinn, idx, val);

    // (2) flow / offsets / sims at scales 1,2,4
    flow_out_kernel<<<(B_N * 64 * 64 + 255) / 256, 256, 0, stream>>>(
        idx, val, out + o_off3, out + o_fl3, out + o_s3, 1);
    flow_out_kernel<<<(B_N * 128 * 128 + 255) / 256, 256, 0, stream>>>(
        idx, val, out + o_off2, out + o_fl2, out + o_s2, 2);
    flow_out_kernel<<<(B_N * 256 * 256 + 255) / 256, 256, 0, stream>>>(
        idx, val, out + o_off1, out + o_fl1, out + o_s1, 4);

    // (3) VGG prefix
    imgnorm_kernel<<<(B_N * 3 * 65536 + 255) / 256, 256, 0, stream>>>(img, vx);
    conv3x3_relu_kernel<3><<<(B_N * 64 * 65536 + 255) / 256, 256, 0, stream>>>(
        vx, w11, b11, out + o_r1, 64, 256);
    conv3x3_relu_kernel<64><<<(B_N * 64 * 65536 + 255) / 256, 256, 0, stream>>>(
        out + o_r1, w12, b12, vt, 64, 256);
    pool2x2_kernel<<<(B_N * 64 * 16384 + 255) / 256, 256, 0, stream>>>(vt, vp, 64, 128);
    conv3x3_relu_kernel<64><<<(B_N * 128 * 16384 + 255) / 256, 256, 0, stream>>>(
        vp, w21, b21, out + o_r2, 128, 128);
    conv3x3_relu_kernel<128><<<(B_N * 128 * 16384 + 255) / 256, 256, 0, stream>>>(
        out + o_r2, w22, b22, vt, 128, 128);
    pool2x2_kernel<<<(B_N * 128 * 4096 + 255) / 256, 256, 0, stream>>>(vt, vp, 128, 64);
    conv3x3_relu_kernel<128><<<(B_N * 256 * 4096 + 255) / 256, 256, 0, stream>>>(
        vp, w31, b31, out + o_r3, 256, 64);
}
